// Copy_Decoder_34608846471828
// MI455X (gfx1250) — compile-verified, matrix operand order checked by reading, not run on hardware
//
#include <hip/hip_runtime.h>
#include <hip/hip_bf16.h>
#include <stdint.h>

// ---------------------------------------------------------------------------
// Problem constants
// ---------------------------------------------------------------------------
#define VSZ   32000
#define ESZ   512
#define HSZ   1024
#define TSLOT 40
#define TPERS 64
#define BSZ   32

typedef __attribute__((ext_vector_type(16))) __bf16 v16bf;
typedef __attribute__((ext_vector_type(8)))  float  v8f;

// ---------------------------------------------------------------------------
// CDNA5 async global->LDS staging (ASYNCcnt-tracked DMA-style copies)
// ---------------------------------------------------------------------------
__device__ __forceinline__ void async_copy_b128(const float* g, const float* l) {
    unsigned lds = (unsigned)(uintptr_t)l;   // low 32 bits of generic LDS addr
    asm volatile("global_load_async_to_lds_b128 %0, %1, off"
                 :: "v"(lds), "v"(g) : "memory");
}

// ---------------------------------------------------------------------------
// Generic C(M,N) = act( A(M,K) @ Bw(N,K)^T + bias ), fp32 in/out, bf16 WMMA.
// Block tile 32x128, 8 waves; wave tile 16x32 (two v_wmma_f32_16x16x32_bf16).
// Double-buffered LDS: async DMA of tile k+1 overlaps WMMA on tile k; each
// wave issues 5 async-load instructions per tile, so "s_wait_asynccnt 5"
// guarantees the previous tile has landed (async loads complete in order).
// grid = (N/128, M/32). K must be a multiple of 32.
// ---------------------------------------------------------------------------
template <int ACT>   // 0 = none, 1 = tanh
__global__ __launch_bounds__(256) void gemm_wmma_bf16(
    const float* __restrict__ A, int lda,
    const float* __restrict__ Bw, int ldb,
    const float* __restrict__ bias,
    float* __restrict__ C, int ldc, int K)
{
    __shared__ float sA[2][32][36];   // +4 pad keeps 16B alignment, avoids conflicts
    __shared__ float sB[2][128][36];  // 46 KB total: fine vs 320 KB/WGP

    const int tid   = threadIdx.x;
    const int lane  = tid & 31;
    const int wave  = tid >> 5;
    const int wm    = wave >> 2;           // 0..1  (M half)
    const int wn    = wave & 3;            // 0..3  (N quarter)
    const int lrow  = lane & 15;
    const int hi    = lane >> 4;           // lane group
    const int khalf = hi << 3;             // 0 or 8 (A/B K striping)
    const int mbase = blockIdx.y * 32;
    const int nbase = blockIdx.x * 128;

    v8f acc0 = {};
    v8f acc1 = {};

    const int ar = tid >> 3, ak = (tid & 7) << 2;   // A stage: 1 x b128 / thread

    auto stage = [&](int buf, int k0) {
        async_copy_b128(&A[(size_t)(mbase + ar) * lda + k0 + ak], &sA[buf][ar][ak]);
#pragma unroll
        for (int i = 0; i < 4; ++i) {               // B stage: 4 x b128 / thread
            int idx = tid + (i << 8);
            int br = idx >> 3, bk = (idx & 7) << 2;
            async_copy_b128(&Bw[(size_t)(nbase + br) * ldb + k0 + bk], &sB[buf][br][bk]);
        }
    };

    const int nt = K >> 5;
    stage(0, 0);

    for (int kt = 0; kt < nt; ++kt) {
        const int cur = kt & 1;
        if (kt + 1 < nt) {
            stage(cur ^ 1, (kt + 1) << 5);
            // speculative L2 prefetch of tile k+2 (dropped if unmapped)
            __builtin_prefetch(&Bw[(size_t)(nbase + (tid >> 1)) * ldb + ((kt + 2) << 5)], 0, 1);
            asm volatile("s_wait_asynccnt 5" ::: "memory");  // tile kt landed
        } else {
            asm volatile("s_wait_asynccnt 0" ::: "memory");  // drain last tile
        }
        __syncthreads();

        // ---- build bf16 fragments (A-layout: lanes0-15 K{0..7,16..23}) ----
        v16bf af, bf0, bf1;
#pragma unroll
        for (int i = 0; i < 8; ++i) {
            af[i]      = (__bf16)sA[cur][wm * 16 + lrow][khalf + i];
            af[8 + i]  = (__bf16)sA[cur][wm * 16 + lrow][16 + khalf + i];
            bf0[i]     = (__bf16)sB[cur][wn * 32 + lrow][khalf + i];
            bf0[8 + i] = (__bf16)sB[cur][wn * 32 + lrow][16 + khalf + i];
            bf1[i]     = (__bf16)sB[cur][wn * 32 + 16 + lrow][khalf + i];
            bf1[8 + i] = (__bf16)sB[cur][wn * 32 + 16 + lrow][16 + khalf + i];
        }
        acc0 = __builtin_amdgcn_wmma_f32_16x16x32_bf16(false, af, false, bf0,
                                                       (short)0, acc0, false, false);
        acc1 = __builtin_amdgcn_wmma_f32_16x16x32_bf16(false, af, false, bf1,
                                                       (short)0, acc1, false, false);
        __syncthreads();   // readers done before this buffer is re-staged
    }

    // ---- epilogue: D layout VGPR r -> M = r + 8*hi, N = lane&15 ----
    const int n0 = nbase + wn * 32 + lrow;
    const int n1 = n0 + 16;
    const float b0 = bias ? bias[n0] : 0.f;
    const float b1 = bias ? bias[n1] : 0.f;
#pragma unroll
    for (int r = 0; r < 8; ++r) {
        int m = mbase + wm * 16 + hi * 8 + r;
        float v0 = acc0[r] + b0;
        float v1 = acc1[r] + b1;
        if (ACT == 1) { v0 = tanhf(v0); v1 = tanhf(v1); }
        C[(size_t)m * ldc + n0] = v0;
        C[(size_t)m * ldc + n1] = v1;
    }
}

// ---------------------------------------------------------------------------
// Attention finish: e[b,t] = v . tanh(hp[b]+ep[t,b]+bias); softmax over t;
// ctx[b] = sum_t a[t] * enc[t,b,:].   One block per batch row.
// ---------------------------------------------------------------------------
__global__ __launch_bounds__(256) void attn_finish_kernel(
    const float* __restrict__ hp,   // (B,H)
    const float* __restrict__ ep,   // (T*B,H)
    const float* __restrict__ bias, // (H)
    const float* __restrict__ vv,   // (H)
    const float* __restrict__ enc,  // (T,B,H)
    float* __restrict__ ctx,        // (B,H)
    int T)
{
    const int b = blockIdx.x, tid = threadIdx.x;
    __shared__ float red[256];
    __shared__ float e[TPERS];
    __shared__ float zsh;

    for (int t = 0; t < T; ++t) {
        float local = 0.f;
        for (int h = tid; h < HSZ; h += 256) {
            float x = hp[b * HSZ + h] + ep[(size_t)(t * BSZ + b) * HSZ + h] + bias[h];
            local += vv[h] * tanhf(x);
        }
        red[tid] = local; __syncthreads();
        for (int s = 128; s > 0; s >>= 1) { if (tid < s) red[tid] += red[tid + s]; __syncthreads(); }
        if (tid == 0) e[t] = red[0];
        __syncthreads();
    }
    if (tid == 0) {
        float m = -3.4e38f;
        for (int t = 0; t < T; ++t) m = fmaxf(m, e[t]);
        float z = 0.f;
        for (int t = 0; t < T; ++t) { e[t] = expf(e[t] - m); z += e[t]; }
        zsh = z;
    }
    __syncthreads();
    const float invz = 1.f / zsh;
    for (int h = tid; h < HSZ; h += 256) {
        float acc = 0.f;
        for (int t = 0; t < T; ++t)
            acc += e[t] * enc[(size_t)(t * BSZ + b) * HSZ + h];
        ctx[b * HSZ + h] = acc * invz;
    }
}

// ---------------------------------------------------------------------------
// Build GRU input (B,2560)=[emb gather | a_ctx | p_ctx]; also fill first 2048
// columns of the projection input xp (B,3072)=[a_ctx | p_ctx | h_new later].
// ---------------------------------------------------------------------------
__global__ void build_inputs_kernel(
    const int* __restrict__ m_t, const float* __restrict__ emb,
    const float* __restrict__ actx, const float* __restrict__ pctx,
    float* __restrict__ xg, float* __restrict__ xp)
{
    int i = blockIdx.x * blockDim.x + threadIdx.x;
    if (i >= BSZ * (ESZ + 2 * HSZ)) return;
    int b = i / (ESZ + 2 * HSZ), c = i % (ESZ + 2 * HSZ);
    float val;
    if (c < ESZ) {
        val = emb[(size_t)m_t[b] * ESZ + c];
    } else if (c < ESZ + HSZ) {
        val = actx[b * HSZ + (c - ESZ)];
        xp[(size_t)b * 3 * HSZ + (c - ESZ)] = val;
    } else {
        val = pctx[b * HSZ + (c - ESZ - HSZ)];
        xp[(size_t)b * 3 * HSZ + HSZ + (c - ESZ - HSZ)] = val;
    }
    xg[i] = val;
}

// ---------------------------------------------------------------------------
// GRU gates: h_new = (1-z)*n + z*h.  Writes h_new into both outputs and into
// xp columns [2048,3072) for the projection GEMM.
// ---------------------------------------------------------------------------
__global__ void gru_gates_kernel(
    const float* __restrict__ gi, const float* __restrict__ gh,
    const float* __restrict__ hold,
    float* __restrict__ out_newhid, float* __restrict__ out_gruout,
    float* __restrict__ xp)
{
    int i = blockIdx.x * blockDim.x + threadIdx.x;
    if (i >= BSZ * HSZ) return;
    int b = i / HSZ, h = i % HSZ;
    size_t r = (size_t)b * 3 * HSZ;
    float ir = gi[r + h], iz = gi[r + HSZ + h], in_ = gi[r + 2 * HSZ + h];
    float hr = gh[r + h], hz = gh[r + HSZ + h], hn  = gh[r + 2 * HSZ + h];
    float rg = 1.f / (1.f + expf(-(ir + hr)));
    float zg = 1.f / (1.f + expf(-(iz + hz)));
    float ng = tanhf(in_ + rg * hn);
    float hv = (1.f - zg) * ng + zg * hold[i];
    out_newhid[i] = hv;
    out_gruout[i] = hv;
    xp[r + 2 * HSZ + h] = hv;
}

// ---------------------------------------------------------------------------
// Copy scores: u_score[b,t] = up[t,b,:] . h_new[b,:]; then rowwise max/exp/sum.
// One block per batch row.
// ---------------------------------------------------------------------------
__global__ __launch_bounds__(256) void copy_scores_kernel(
    const float* __restrict__ up,   // (TSLOT*B, H), already tanh'd
    const float* __restrict__ xp,   // h_new at xp[b*3072 + 2048 + h]
    float* __restrict__ uexp,       // (B, TSLOT)
    float* __restrict__ Ssum,       // (B)
    float* __restrict__ umax)       // (B)
{
    const int b = blockIdx.x, tid = threadIdx.x;
    __shared__ float red[256];
    __shared__ float us[TSLOT];
    for (int t = 0; t < TSLOT; ++t) {
        float local = 0.f;
        for (int h = tid; h < HSZ; h += 256)
            local += up[(size_t)(t * BSZ + b) * HSZ + h] * xp[(size_t)b * 3 * HSZ + 2 * HSZ + h];
        red[tid] = local; __syncthreads();
        for (int s = 128; s > 0; s >>= 1) { if (tid < s) red[tid] += red[tid + s]; __syncthreads(); }
        if (tid == 0) us[t] = red[0];
        __syncthreads();
    }
    if (tid == 0) {
        float m = -3.4e38f;
        for (int t = 0; t < TSLOT; ++t) m = fmaxf(m, us[t]);
        float S = 0.f;
        for (int t = 0; t < TSLOT; ++t) {
            float e = expf(us[t] - m);
            uexp[b * TSLOT + t] = e;
            S += e;
        }
        Ssum[b] = S;
        umax[b] = m;
    }
}

// ---------------------------------------------------------------------------
// w buffer: w[b,v'] = 1e-10 * S_b  (+ scatter of val*u_exp).  u_copy = log(w)+u_max.
// ---------------------------------------------------------------------------
__global__ void init_w_kernel(const float* __restrict__ Ssum, float* __restrict__ wbuf)
{
    int i = blockIdx.x * blockDim.x + threadIdx.x;
    if (i >= BSZ * (VSZ + TSLOT)) return;
    int b = i / (VSZ + TSLOT);
    wbuf[i] = 1e-10f * Ssum[b];
}

__global__ void scatter_w_kernel(const int* __restrict__ slot_np,
                                 const float* __restrict__ uexp,
                                 float* __restrict__ wbuf)
{
    int i = blockIdx.x * blockDim.x + threadIdx.x;
    if (i >= TSLOT * BSZ) return;
    int t = i / BSZ, b = i % BSZ;
    int v = slot_np[t * BSZ + b];
    float ue = uexp[b * TSLOT + t];
    if (v == 2)
        atomicAdd(&wbuf[(size_t)b * (VSZ + TSLOT) + VSZ + t], (5.0f - 1e-10f) * ue);
    else if (v != 0)
        atomicAdd(&wbuf[(size_t)b * (VSZ + TSLOT) + v], ue);
}

// ---------------------------------------------------------------------------
// Final fused softmax over concat(gen_score, u_copy) and combine:
// proba[v<V] = softmax_gen[v] + softmax_copy[v];  proba[V+t] = softmax_copy[V+t].
// exp(u_copy - m) = w * exp(u_max - m), so no log/exp round trip needed.
// One block per batch row, 3 passes.
// ---------------------------------------------------------------------------
__global__ __launch_bounds__(256) void finalize_kernel(
    const float* __restrict__ gen,   // (B, V)
    const float* __restrict__ wbuf,  // (B, V+TSLOT)
    const float* __restrict__ umax,  // (B)
    float* __restrict__ out)         // (B, V+TSLOT)
{
    const int b = blockIdx.x, tid = threadIdx.x;
    const int W = VSZ + TSLOT;
    __shared__ float red[256];
    __shared__ float msh, zsh;
    const float* g = gen + (size_t)b * VSZ;
    const float* w = wbuf + (size_t)b * W;

    float m1 = -3.4e38f;
    for (int v = tid; v < VSZ; v += 256) m1 = fmaxf(m1, g[v]);
    red[tid] = m1; __syncthreads();
    for (int s = 128; s > 0; s >>= 1) { if (tid < s) red[tid] = fmaxf(red[tid], red[tid + s]); __syncthreads(); }
    float m1r = red[0]; __syncthreads();

    float m2 = 0.f;
    for (int v = tid; v < W; v += 256) m2 = fmaxf(m2, w[v]);
    red[tid] = m2; __syncthreads();
    for (int s = 128; s > 0; s >>= 1) { if (tid < s) red[tid] = fmaxf(red[tid], red[tid + s]); __syncthreads(); }
    if (tid == 0) msh = fmaxf(m1r, logf(red[0]) + umax[b]);
    __syncthreads();
    const float m = msh;
    const float e0 = expf(umax[b] - m);

    float z = 0.f;
    for (int v = tid; v < VSZ; v += 256) z += expf(g[v] - m);
    for (int v = tid; v < W; v += 256) z += w[v] * e0;
    red[tid] = z; __syncthreads();
    for (int s = 128; s > 0; s >>= 1) { if (tid < s) red[tid] += red[tid + s]; __syncthreads(); }
    if (tid == 0) zsh = red[0];
    __syncthreads();
    const float invZ = 1.f / zsh;

    float* ob = out + (size_t)b * W;
    for (int v = tid; v < VSZ; v += 256)
        ob[v] = (expf(g[v] - m) + w[v] * e0) * invZ;
    for (int v = VSZ + tid; v < W; v += 256)
        ob[v] = w[v] * e0 * invZ;
}

// ---------------------------------------------------------------------------
// Host launcher
// ---------------------------------------------------------------------------
extern "C" void kernel_launch(void* const* d_in, const int* in_sizes, int n_in,
                              void* d_out, int out_size, void* d_ws, size_t ws_size,
                              hipStream_t stream) {
    const float* slot_enc = (const float*)d_in[0];   // (40,32,1024)
    const int*   slot_np  = (const int*)  d_in[1];   // (40,32)
    const float* pers_enc = (const float*)d_in[2];   // (64,32,1024)
    const int*   m_t      = (const int*)  d_in[3];   // (1,32)
    const float* last_h   = (const float*)d_in[4];   // (1,32,1024)
    const float* emb      = (const float*)d_in[5];   // (32000,512)
    const float* aW = (const float*)d_in[6];         // (1024,2048)
    const float* ab = (const float*)d_in[7];
    const float* av = (const float*)d_in[8];
    const float* pW = (const float*)d_in[9];
    const float* pb = (const float*)d_in[10];
    const float* pv = (const float*)d_in[11];
    const float* Wih = (const float*)d_in[12];       // (3072,2560)
    const float* Whh = (const float*)d_in[13];       // (3072,1024)
    const float* bih = (const float*)d_in[14];
    const float* bhh = (const float*)d_in[15];
    const float* projW = (const float*)d_in[16];     // (32000,3072)
    const float* projb = (const float*)d_in[17];
    const float* c2W = (const float*)d_in[18];       // (1024,1024)
    const float* c2b = (const float*)d_in[19];

    float* ws = (float*)d_ws;
    size_t o = 0;
    float* hpA  = ws + o; o += BSZ * HSZ;
    float* hpP  = ws + o; o += BSZ * HSZ;
    float* epA  = ws + o; o += (size_t)TSLOT * BSZ * HSZ;
    float* epP  = ws + o; o += (size_t)TPERS * BSZ * HSZ;
    float* actx = ws + o; o += BSZ * HSZ;
    float* pctx = ws + o; o += BSZ * HSZ;
    float* xg   = ws + o; o += BSZ * (ESZ + 2 * HSZ);
    float* xp   = ws + o; o += BSZ * 3 * HSZ;
    float* gi   = ws + o; o += BSZ * 3 * HSZ;
    float* gh   = ws + o; o += BSZ * 3 * HSZ;
    float* gen  = ws + o; o += (size_t)BSZ * VSZ;
    float* up   = ws + o; o += (size_t)TSLOT * BSZ * HSZ;
    float* uexp = ws + o; o += BSZ * TSLOT;
    float* Ssum = ws + o; o += BSZ;
    float* umax = ws + o; o += BSZ;
    float* wbuf = ws + o; o += (size_t)BSZ * (VSZ + TSLOT);

    float* out_proba  = (float*)d_out;                       // (32, 32040)
    float* out_newhid = out_proba + (size_t)BSZ * (VSZ + TSLOT);
    float* out_gruout = out_newhid + BSZ * HSZ;

    // 1) hidden projections (hidden @ W[:, :H]^T): M=32, N=1024, K=1024
    gemm_wmma_bf16<0><<<dim3(HSZ / 128, 1), 256, 0, stream>>>(last_h, HSZ, aW, 2 * HSZ, nullptr, hpA, HSZ, HSZ);
    gemm_wmma_bf16<0><<<dim3(HSZ / 128, 1), 256, 0, stream>>>(last_h, HSZ, pW, 2 * HSZ, nullptr, hpP, HSZ, HSZ);
    // 2) encoder projections (enc @ W[:, H:]^T): M=T*B, N=1024, K=1024
    gemm_wmma_bf16<0><<<dim3(HSZ / 128, TSLOT * BSZ / 32), 256, 0, stream>>>(slot_enc, HSZ, aW + HSZ, 2 * HSZ, nullptr, epA, HSZ, HSZ);
    gemm_wmma_bf16<0><<<dim3(HSZ / 128, TPERS * BSZ / 32), 256, 0, stream>>>(pers_enc, HSZ, pW + HSZ, 2 * HSZ, nullptr, epP, HSZ, HSZ);
    // 3) attention energies + softmax + context
    attn_finish_kernel<<<BSZ, 256, 0, stream>>>(hpA, epA, ab, av, slot_enc, actx, TSLOT);
    attn_finish_kernel<<<BSZ, 256, 0, stream>>>(hpP, epP, pb, pv, pers_enc, pctx, TPERS);
    // 4) GRU input assembly + gi/gh GEMMs + gates
    build_inputs_kernel<<<(BSZ * (ESZ + 2 * HSZ) + 255) / 256, 256, 0, stream>>>(m_t, emb, actx, pctx, xg, xp);
    gemm_wmma_bf16<0><<<dim3(3 * HSZ / 128, 1), 256, 0, stream>>>(xg, ESZ + 2 * HSZ, Wih, ESZ + 2 * HSZ, bih, gi, 3 * HSZ, ESZ + 2 * HSZ);
    gemm_wmma_bf16<0><<<dim3(3 * HSZ / 128, 1), 256, 0, stream>>>(last_h, HSZ, Whh, HSZ, bhh, gh, 3 * HSZ, HSZ);
    gru_gates_kernel<<<(BSZ * HSZ + 255) / 256, 256, 0, stream>>>(gi, gh, last_h, out_newhid, out_gruout, xp);
    // 5) vocab projection (the 393 MB GEMM): M=32, N=32000, K=3072
    gemm_wmma_bf16<0><<<dim3(VSZ / 128, 1), 256, 0, stream>>>(xp, 3 * HSZ, projW, 3 * HSZ, projb, gen, VSZ, 3 * HSZ);
    // 6) copy attention: u = tanh(slot_enc @ copy2_W^T + b), scores, scatter
    gemm_wmma_bf16<1><<<dim3(HSZ / 128, TSLOT * BSZ / 32), 256, 0, stream>>>(slot_enc, HSZ, c2W, HSZ, c2b, up, HSZ, HSZ);
    copy_scores_kernel<<<BSZ, 256, 0, stream>>>(up, xp, uexp, Ssum, umax);
    init_w_kernel<<<(BSZ * (VSZ + TSLOT) + 255) / 256, 256, 0, stream>>>(Ssum, wbuf);
    scatter_w_kernel<<<(TSLOT * BSZ + 255) / 256, 256, 0, stream>>>(slot_np, uexp, wbuf);
    // 7) fused final softmax + combine
    finalize_kernel<<<BSZ, 256, 0, stream>>>(gen, wbuf, umax, out_proba);
}
